// AggregationDiscriminationLoss_30958124270216
// MI455X (gfx1250) — matrix operand order repaired and checked by reading, not compiled, not run on hardware
//
#include <hip/hip_runtime.h>

// CDNA5 / gfx1250. Segmented reductions (M=16 segments) expressed as
// one-hot WMMA matmuls:  D[16x16] += onehot(ids)[16x32] * values[32x16]
// using v_wmma_f32_16x16x32_f16 with f32 accumulation.
// B-tile lives in LDS padded to all 16 columns (constant count/zero columns
// pre-initialized) so the inner loop has NO divergent branches: every lane
// issues two ds_load_b128 and one v_wmma per 32-pixel chunk.

typedef __attribute__((ext_vector_type(16))) _Float16 v16h;
typedef __attribute__((ext_vector_type(4)))  _Float16 v4h;
typedef __attribute__((ext_vector_type(8)))  float    v8f;

#define MSEG   16
#define NB     8
#define PIX    409600          /* 640*640 */
#define BLKPIX 128
#define NBLK   (PIX / BLKPIX)  /* 3200 */
#define DELTA_AGG 0.5f
#define DELTA_DIS 3.0f

__device__ __forceinline__ void atomAddF(float* p, float v) {
    __hip_atomic_fetch_add(p, v, __ATOMIC_RELAXED, __HIP_MEMORY_SCOPE_AGENT);
}

// ---------------------------------------------------------------------------
// Kernel 1: segment-sum sim (4 ch) and counts by `kern` id via WMMA.
// Ksum layout: [N][16][5]  (cols 0..3 = sum of sim channels, col 4 = cnt_k)
// ---------------------------------------------------------------------------
__global__ void __launch_bounds__(256)
k_kern_segsum(const float* __restrict__ preds, const int* __restrict__ targets,
              float* __restrict__ Ksum)
{
    // Per-wave B tile: [16 cols][128 pixels] f16. Cols 0..3 = sim channels,
    // col 4 = 1.0 (count), cols 5..15 = 0.
    __shared__ _Float16      bt_s[8][16 * BLKPIX];
    __shared__ unsigned char id_s[8][BLKPIX];

    const int n    = blockIdx.y;
    const int wv   = threadIdx.x >> 5;
    const int lane = threadIdx.x & 31;
    const int m    = lane & 15;   // A: row M / B: col N / D: col N
    const int hi   = lane >> 4;

    const float* simb = preds   + (size_t)(n * 6 + 2) * PIX;
    const int*   kern = targets + (size_t)(n * 2 + 1) * PIX;

    _Float16*      bt  = bt_s[wv];
    unsigned char* idb = id_s[wv];

    // One-time init of constant columns 4..15 (wave-private; DS is in-order).
    for (int i = lane; i < (12 * BLKPIX) / 4; i += 32) {
        const int off = i * 4;                  // half offset within cols 4..15
        const int col = 4 + off / BLKPIX;
        const _Float16 v = (col == 4) ? (_Float16)1.0f : (_Float16)0.0f;
        v4h h; h[0] = v; h[1] = v; h[2] = v; h[3] = v;
        *(v4h*)&bt[4 * BLKPIX + off] = h;
    }

    v8f acc = {};

    const int wstride = gridDim.x * 8;
    for (int blk = blockIdx.x * 8 + wv; blk < NBLK; blk += wstride) {
        const int base = blk * BLKPIX;
        const int p    = base + lane * 4;

        // Prefetch next iteration's lines (global_prefetch_b8).
        const int nb = blk + wstride;
        if (nb < NBLK) {
            const float* np = simb + (size_t)nb * BLKPIX + lane * 4;
#pragma unroll
            for (int c = 0; c < 4; ++c)
                __builtin_prefetch(np + (size_t)c * PIX, 0, 1);
            __builtin_prefetch(kern + (size_t)nb * BLKPIX + lane * 4, 0, 1);
        }

        // Stage 128 pixels: 4 channels -> f16 tile [c][k]; kern ids -> bytes.
#pragma unroll
        for (int c = 0; c < 4; ++c) {
            const float4 v = *(const float4*)(simb + (size_t)c * PIX + p);
            v4h h;
            h[0] = (_Float16)v.x; h[1] = (_Float16)v.y;
            h[2] = (_Float16)v.z; h[3] = (_Float16)v.w;
            *(v4h*)&bt[c * BLKPIX + lane * 4] = h;
        }
        {
            const int4 kv = *(const int4*)(kern + p);
            const unsigned pk = (unsigned)(kv.x & 0xff)
                              | ((unsigned)(kv.y & 0xff) << 8)
                              | ((unsigned)(kv.z & 0xff) << 16)
                              | ((unsigned)(kv.w & 0xff) << 24);
            *(unsigned*)&idb[lane * 4] = pk;
        }
        // DS ops from the same wave complete in order; compiler inserts s_wait_dscnt.

        // 4 chunks of 32 pixels -> 4 WMMAs, branch-free.
#pragma unroll
        for (int ch = 0; ch < 4; ++ch) {
            const int cb    = ch * 32;
            const int kbase = cb + hi * 16;
            const unsigned long long idlo = *(const unsigned long long*)&idb[kbase];
            const unsigned long long idhi = *(const unsigned long long*)&idb[kbase + 8];
            v16h a;  // onehot(kern[k] == m), A-layout K set {base..base+15} contiguous
#pragma unroll
            for (int j = 0; j < 8; ++j) {
                a[j]     = ((int)((idlo >> (8 * j)) & 0xffull) == m) ? (_Float16)1.0f : (_Float16)0.0f;
                a[j + 8] = ((int)((idhi >> (8 * j)) & 0xffull) == m) ? (_Float16)1.0f : (_Float16)0.0f;
            }
            // Column n=m of B: rows K = cb+16*hi .. +15 contiguous (2x ds_load_b128).
            const v16h b = *(const v16h*)&bt[m * BLKPIX + cb + hi * 16];

            acc = __builtin_amdgcn_wmma_f32_16x16x32_f16(
                false, a, false, b, (short)0, acc, false, false);
        }
    }

    // D: lane holds col m, rows j + 8*hi.
    if (m < 5) {
#pragma unroll
        for (int j = 0; j < 8; ++j) {
            const int row = j + 8 * hi;
            atomAddF(&Ksum[(n * MSEG + row) * 5 + m], acc[j]);
        }
    }
}

// ---------------------------------------------------------------------------
// Kernel 2: G[n][m][c] = Ksum[n][m][c] / max(cnt_k, 1)
// ---------------------------------------------------------------------------
__global__ void k_compute_G(const float* __restrict__ Ksum, float* __restrict__ G)
{
    const int t = threadIdx.x;            // 0..127 == n*16 + m
    if (t < NB * MSEG) {
        const float inv = 1.0f / fmaxf(Ksum[t * 5 + 4], 1.0f);
#pragma unroll
        for (int c = 0; c < 4; ++c) G[t * 4 + c] = Ksum[t * 5 + c] * inv;
    }
}

// ---------------------------------------------------------------------------
// Kernel 3: per-pixel aggregation loss, segment-summed by `text` id via WMMA.
// Lsum layout: [N][16][2]  (col 0 = sum of l, col 1 = cnt_t)
// ---------------------------------------------------------------------------
__global__ void __launch_bounds__(256)
k_agg_segsum(const float* __restrict__ preds, const int* __restrict__ targets,
             const float* __restrict__ G, float* __restrict__ Lsum)
{
    __shared__ float         Gl[MSEG][4];
    // Per-wave B tile: [16 cols][128 pixels]. Col 0 = l, col 1 = 1.0, rest 0.
    __shared__ _Float16      lt_s[8][16 * BLKPIX];
    __shared__ unsigned char id_s[8][BLKPIX];

    const int n    = blockIdx.y;
    const int wv   = threadIdx.x >> 5;
    const int lane = threadIdx.x & 31;
    const int m    = lane & 15;
    const int hi   = lane >> 4;

    if (threadIdx.x < MSEG * 4)
        ((float*)Gl)[threadIdx.x] = G[n * MSEG * 4 + threadIdx.x];
    __syncthreads();  // once, before the loop (uniform across block)

    const float* simb = preds   + (size_t)(n * 6 + 2) * PIX;
    const int*   text = targets + (size_t)(n * 2) * PIX;

    _Float16*      lt  = lt_s[wv];
    unsigned char* idb = id_s[wv];

    // One-time init of constant columns 1..15.
    for (int i = lane; i < (15 * BLKPIX) / 4; i += 32) {
        const int off = i * 4;
        const int col = 1 + off / BLKPIX;
        const _Float16 v = (col == 1) ? (_Float16)1.0f : (_Float16)0.0f;
        v4h h; h[0] = v; h[1] = v; h[2] = v; h[3] = v;
        *(v4h*)&lt[BLKPIX + off] = h;
    }

    v8f acc = {};

    const int wstride = gridDim.x * 8;
    for (int blk = blockIdx.x * 8 + wv; blk < NBLK; blk += wstride) {
        const int base = blk * BLKPIX;
        const int p    = base + lane * 4;

        const int nb = blk + wstride;
        if (nb < NBLK) {
            const float* np = simb + (size_t)nb * BLKPIX + lane * 4;
#pragma unroll
            for (int c = 0; c < 4; ++c)
                __builtin_prefetch(np + (size_t)c * PIX, 0, 1);
            __builtin_prefetch(text + (size_t)nb * BLKPIX + lane * 4, 0, 1);
        }

        const float4 c0 = *(const float4*)(simb + (size_t)0 * PIX + p);
        const float4 c1 = *(const float4*)(simb + (size_t)1 * PIX + p);
        const float4 c2 = *(const float4*)(simb + (size_t)2 * PIX + p);
        const float4 c3 = *(const float4*)(simb + (size_t)3 * PIX + p);
        const int4   tv = *(const int4*)(text + p);

        const float s0[4] = {c0.x, c0.y, c0.z, c0.w};
        const float s1[4] = {c1.x, c1.y, c1.z, c1.w};
        const float s2[4] = {c2.x, c2.y, c2.z, c2.w};
        const float s3[4] = {c3.x, c3.y, c3.z, c3.w};
        const int   td[4] = {tv.x, tv.y, tv.z, tv.w};

        v4h lh;
#pragma unroll
        for (int j = 0; j < 4; ++j) {
            const float4 g = *(const float4*)&Gl[td[j] & 15][0];
            const float dx = s0[j] - g.x, dy = s1[j] - g.y;
            const float dz = s2[j] - g.z, dw = s3[j] - g.w;
            const float d  = sqrtf(dx * dx + dy * dy + dz * dz + dw * dw) - DELTA_AGG;
            const float dp = fmaxf(d, 0.0f);
            lh[j] = (_Float16)__logf(__fmaf_rn(dp, dp, 1.0f));
        }
        *(v4h*)&lt[lane * 4] = lh;
        {
            const unsigned pk = (unsigned)(td[0] & 0xff)
                              | ((unsigned)(td[1] & 0xff) << 8)
                              | ((unsigned)(td[2] & 0xff) << 16)
                              | ((unsigned)(td[3] & 0xff) << 24);
            *(unsigned*)&idb[lane * 4] = pk;
        }

#pragma unroll
        for (int ch = 0; ch < 4; ++ch) {
            const int cb    = ch * 32;
            const int kbase = cb + hi * 16;
            const unsigned long long idlo = *(const unsigned long long*)&idb[kbase];
            const unsigned long long idhi = *(const unsigned long long*)&idb[kbase + 8];
            v16h a;
#pragma unroll
            for (int j = 0; j < 8; ++j) {
                a[j]     = ((int)((idlo >> (8 * j)) & 0xffull) == m) ? (_Float16)1.0f : (_Float16)0.0f;
                a[j + 8] = ((int)((idhi >> (8 * j)) & 0xffull) == m) ? (_Float16)1.0f : (_Float16)0.0f;
            }
            const v16h b = *(const v16h*)&lt[m * BLKPIX + cb + hi * 16];

            acc = __builtin_amdgcn_wmma_f32_16x16x32_f16(
                false, a, false, b, (short)0, acc, false, false);
        }
    }

    if (m < 2) {
#pragma unroll
        for (int j = 0; j < 8; ++j) {
            const int row = j + 8 * hi;
            atomAddF(&Lsum[(n * MSEG + row) * 2 + m], acc[j]);
        }
    }
}

// ---------------------------------------------------------------------------
// Kernel 4: finalize. One wave per batch. out = [agg(8), dis(8)].
// ---------------------------------------------------------------------------
__global__ void __launch_bounds__(256)
k_finalize(const float* __restrict__ Ksum, const float* __restrict__ Lsum,
           const float* __restrict__ G, float* __restrict__ out)
{
    const int n    = threadIdx.x >> 5;
    const int lane = threadIdx.x & 31;

    float cntt = 0.0f, ls = 0.0f;
    bool  valid = false;
    if (lane < MSEG) {
        const float cntk = Ksum[(n * MSEG + lane) * 5 + 4];
        cntt  = Lsum[(n * MSEG + lane) * 2 + 1];
        ls    = Lsum[(n * MSEG + lane) * 2 + 0];
        valid = (cntk > 0.5f) && (cntt > 0.5f) && (lane >= 1);
    }
    const unsigned long long vm = __ballot(valid);
    const int nv = __popcll(vm);

    float aggc = valid ? ls / fmaxf(cntt, 1.0f) : 0.0f;
#pragma unroll
    for (int off = 16; off > 0; off >>= 1) aggc += __shfl_down(aggc, off);

    float lpsum = 0.0f;
    for (int t = lane; t < MSEG * MSEG; t += 32) {
        const int m1 = t >> 4, m2 = t & 15;
        if (m1 != m2 && ((vm >> m1) & 1ull) && ((vm >> m2) & 1ull)) {
            float sq = 0.0f;
#pragma unroll
            for (int c = 0; c < 4; ++c) {
                const float d = G[(n * MSEG + m1) * 4 + c] - G[(n * MSEG + m2) * 4 + c];
                sq += d * d;
            }
            const float r = fmaxf(DELTA_DIS - sqrtf(sq), 0.0f);
            lpsum += __logf(r * r + 1.0f);
        }
    }
#pragma unroll
    for (int off = 16; off > 0; off >>= 1) lpsum += __shfl_down(lpsum, off);

    if (lane == 0) {
        out[n]      = aggc / fmaxf((float)nv, 1.0f);
        out[NB + n] = (nv > 1)
                        ? 0.5f * lpsum / fmaxf((float)(nv * (nv - 1)), 1.0f)
                        : 0.0f;
    }
}

// ---------------------------------------------------------------------------
extern "C" void kernel_launch(void* const* d_in, const int* in_sizes, int n_in,
                              void* d_out, int out_size, void* d_ws, size_t ws_size,
                              hipStream_t stream) {
    const float* preds   = (const float*)d_in[0];
    const int*   targets = (const int*)d_in[1];
    float*       out     = (float*)d_out;

    float* Ksum = (float*)d_ws;            // 8*16*5 = 640 floats (atomic accum)
    float* Lsum = (float*)d_ws + 640;      // 8*16*2 = 256 floats (atomic accum)
    float* G    = (float*)d_ws + 1024;     // 8*16*4 = 512 floats

    // Zero the atomic accumulators (Ksum + Lsum) every call.
    hipMemsetAsync(d_ws, 0, 896 * sizeof(float), stream);

    const dim3 blk(256);
    k_kern_segsum<<<dim3(64, NB), blk, 0, stream>>>(preds, targets, Ksum);
    k_compute_G  <<<dim3(1), dim3(128), 0, stream>>>(Ksum, G);
    k_agg_segsum <<<dim3(64, NB), blk, 0, stream>>>(preds, targets, G, Lsum);
    k_finalize   <<<dim3(1), blk, 0, stream>>>(Ksum, Lsum, G, out);
}
